// Qwen3Attention_50706383896831
// MI455X (gfx1250) — matrix operand order repaired and checked by reading, not compiled, not run on hardware
//
#include <hip/hip_runtime.h>
#include <hip/hip_bf16.h>
#include <math.h>
#include <stdint.h>

// ---------------------------------------------------------------------------
// Qwen3 attention block for MI455X (gfx1250), wave32, bf16 WMMA everywhere.
// Round 3: branch-free double-buffered async GEMM pipeline (peeled epilogue).
// ---------------------------------------------------------------------------

typedef __bf16 bf16_t;
typedef __attribute__((ext_vector_type(16))) __bf16 v16bf;
typedef __attribute__((ext_vector_type(8)))  __bf16 v8bf;
typedef __attribute__((ext_vector_type(4)))  __bf16 v4bf;
typedef __attribute__((ext_vector_type(8)))  float  v8f;

#define T_SEQ   2048
#define HIDDEN  4096
#define NH      32
#define NKV     8
#define HD      128

static __device__ inline v8f vzero8() {
    v8f z;
#pragma unroll
    for (int i = 0; i < 8; ++i) z[i] = 0.0f;
    return z;
}

#define WMMA_BF16(a, b, c) \
    __builtin_amdgcn_wmma_f32_16x16x32_bf16(false, (a), false, (b), (short)0, (c), false, false)

// ---- async global->LDS (CDNA5, tracked by ASYNCcnt) -----------------------
// Generic pointers to __shared__ carry the LDS offset in the low 32 bits
// (ISA 10.2, LDS aperture).
static __device__ inline uint32_t lds_off(const void* p) {
    return (uint32_t)(uintptr_t)p;
}
static __device__ inline void async_load_b128(uint32_t lds, const void* gaddr) {
    asm volatile("global_load_async_to_lds_b128 %0, %1, off"
                 :: "v"(lds), "v"(gaddr) : "memory");
}
static __device__ inline void wait_async0() {
    asm volatile("s_wait_asynccnt 0x0" ::: "memory");
}
static __device__ inline void wait_async4() {   // allow 4 in-flight (next stage)
    asm volatile("s_wait_asynccnt 0x4" ::: "memory");
}

// A fragment (16x32 bf16, ISA 7.12.2 layout): lane = R + 16*g, R=lane&15.
// elems 0-7  = src[(row0+R)*ld + k0 + g*8 + 0..7]
// elems 8-15 = src[(row0+R)*ld + k0 + g*8 + 16..23]
static __device__ inline v16bf load_frag_a(const bf16_t* __restrict__ src, int ld,
                                           int row0, int k0) {
    int l = threadIdx.x & 31;
    int R = l & 15, g = l >> 4;
    const bf16_t* p = src + (size_t)(row0 + R) * ld + k0 + g * 8;
    v8bf lo = *(const v8bf*)(p);
    v8bf hi = *(const v8bf*)(p + 16);
    v16bf r;
#pragma unroll
    for (int i = 0; i < 8; ++i) { r[i] = lo[i]; r[i + 8] = hi[i]; }
    return r;
}

// B fragment (32x16 bf16), sourced from row-major [N, K] storage (B = W^T):
// lane = n + 16*g holds B[k0+g*8 .. +7, k0+16+g*8 .. +7][n0+n] (same k order as A).
static __device__ inline v16bf load_frag_b(const bf16_t* __restrict__ src, int ld,
                                           int n0, int k0) {
    int l = threadIdx.x & 31;
    int n = l & 15, g = l >> 4;
    const bf16_t* p = src + (size_t)(n0 + n) * ld + k0 + g * 8;
    v8bf lo = *(const v8bf*)(p);
    v8bf hi = *(const v8bf*)(p + 16);
    v16bf r;
#pragma unroll
    for (int i = 0; i < 8; ++i) { r[i] = lo[i]; r[i + 8] = hi[i]; }
    return r;
}

// ---------------------------------------------------------------------------
// elementwise fp32 -> bf16 cast (vectorized by 4; all sizes here are %4==0)
// ---------------------------------------------------------------------------
__global__ void cvt_f32_to_bf16(const float* __restrict__ in,
                                bf16_t* __restrict__ out, int n4) {
    int i = blockIdx.x * blockDim.x + threadIdx.x;
    if (i < n4) {
        float4 v = ((const float4*)in)[i];
        v4bf b;
        b[0] = (bf16_t)v.x; b[1] = (bf16_t)v.y;
        b[2] = (bf16_t)v.z; b[3] = (bf16_t)v.w;
        ((v4bf*)out)[i] = b;
    }
}

// ---------------------------------------------------------------------------
// GEMM: C[M,N] f32 = A[M,K] bf16 (row major) x B[N,K] bf16 (row major, i.e.
// y = A * B^T).  BM=BN=128, BK=32, 8 waves, double-buffered async staging,
// branch-free steady state (last iteration peeled).
// ---------------------------------------------------------------------------
#define BM 128
#define BN 128
#define BK 32

__global__ void __launch_bounds__(256)
gemm_bf16_nt(const bf16_t* __restrict__ A, const bf16_t* __restrict__ B,
             float* __restrict__ C, int M, int N, int K) {
    __shared__ __align__(16) bf16_t sA[2][BM * BK];
    __shared__ __align__(16) bf16_t sB[2][BN * BK];

    const int bm = blockIdx.y * BM;
    const int bn = blockIdx.x * BN;
    const int t  = threadIdx.x;
    const int wave = t >> 5;
    const int lane = t & 31;
    const int wm = (wave & 1) * 64;   // 2 row groups of 64
    const int wn = (wave >> 1) * 32;  // 4 col groups of 32

    v8f acc[4][2];
#pragma unroll
    for (int i = 0; i < 4; ++i)
#pragma unroll
        for (int j = 0; j < 2; ++j) acc[i][j] = vzero8();

    const int lrow = t >> 1;           // 0..127
    const int lcol = (t & 1) * 16;     // 0 or 16
    const bf16_t* ga = A + (size_t)(bm + lrow) * K + lcol;
    const bf16_t* gb = B + (size_t)(bn + lrow) * K + lcol;

    auto stage = [&](int buf, int k0) {
        async_load_b128(lds_off(&sA[buf][lrow * BK + lcol]),     ga + k0);
        async_load_b128(lds_off(&sA[buf][lrow * BK + lcol + 8]), ga + k0 + 8);
        async_load_b128(lds_off(&sB[buf][lrow * BK + lcol]),     gb + k0);
        async_load_b128(lds_off(&sB[buf][lrow * BK + lcol + 8]), gb + k0 + 8);
    };
    auto compute = [&](int buf) {
        v16bf bfrag[2];
#pragma unroll
        for (int j = 0; j < 2; ++j)
            bfrag[j] = load_frag_b(sB[buf], BK, wn + j * 16, 0);
#pragma unroll
        for (int i = 0; i < 4; ++i) {
            v16bf afrag = load_frag_a(sA[buf], BK, wm + i * 16, 0);
            acc[i][0] = WMMA_BF16(afrag, bfrag[0], acc[i][0]);
            acc[i][1] = WMMA_BF16(afrag, bfrag[1], acc[i][1]);
        }
    };

    const int NT = K / BK;             // always >= 2 here
    stage(0, 0);
    for (int kt = 0; kt < NT - 1; ++kt) {
        const int buf = kt & 1;
        stage(buf ^ 1, (kt + 1) * BK); // next tile in flight
        wait_async4();                 // current tile landed
        __syncthreads();
        compute(buf);
        __syncthreads();               // buf is re-staged next iteration
    }
    wait_async0();
    __syncthreads();
    compute((NT - 1) & 1);

    // C/D layout: VGPR r, lanes 0-15 -> (M=r, N=lane); lanes 16-31 -> (M=8+r).
    const int cn = lane & 15;
    const int hm = (lane >> 4) * 8;
#pragma unroll
    for (int i = 0; i < 4; ++i)
#pragma unroll
        for (int j = 0; j < 2; ++j)
#pragma unroll
            for (int r = 0; r < 8; ++r)
                C[(size_t)(bm + wm + i * 16 + hm + r) * N + (bn + wn + j * 16 + cn)] =
                    acc[i][j][r];
}

// ---------------------------------------------------------------------------
// Fused per-head RMSNorm + RoPE, fp32 in -> bf16 out.
// One wave32 per (t, head) row of 128 elements (4 per lane).
// ---------------------------------------------------------------------------
__global__ void __launch_bounds__(256)
rmsnorm_rope_kernel(const float* __restrict__ in, const float* __restrict__ w,
                    const int* __restrict__ positions, bf16_t* __restrict__ out,
                    int heads, int rows) {
    const int row = blockIdx.x * 8 + (threadIdx.x >> 5);
    if (row >= rows) return;
    const int lane = threadIdx.x & 31;
    const int tpos = row / heads;

    float4 x = *(const float4*)(in + (size_t)row * HD + lane * 4);
    float ss = x.x * x.x + x.y * x.y + x.z * x.z + x.w * x.w;
#pragma unroll
    for (int m = 1; m < 32; m <<= 1) ss += __shfl_xor(ss, m, 32);
    const float rn = rsqrtf(ss * (1.0f / HD) + 1e-6f);

    float4 wv = *(const float4*)(w + lane * 4);
    x.x *= rn * wv.x; x.y *= rn * wv.y; x.z *= rn * wv.z; x.w *= rn * wv.w;

    // rotation partner: d ^ 64  <=>  lane ^ 16
    float ox = __shfl_xor(x.x, 16, 32);
    float oy = __shfl_xor(x.y, 16, 32);
    float oz = __shfl_xor(x.z, 16, 32);
    float ow = __shfl_xor(x.w, 16, 32);
    const float sign = (lane < 16) ? -1.0f : 1.0f;  // d<64: -x[d+64], else +x[d-64]

    const float pos = (float)positions[tpos];
    const float NLT = -13.815510557964274f / 64.0f;  // -ln(1e6)/64
    float xi[4] = {x.x, x.y, x.z, x.w};
    float oi[4] = {ox, oy, oz, ow};
    bf16_t* orow = out + (size_t)row * HD + lane * 4;
    v4bf ov;
#pragma unroll
    for (int c = 0; c < 4; ++c) {
        int i = (lane * 4 + c) >> 1;
        float ang = pos * __expf(NLT * (float)i);
        float cs = __cosf(ang), sn = __sinf(ang);
        ov[c] = (bf16_t)(xi[c] * cs + sign * oi[c] * sn);
    }
    *(v4bf*)orow = ov;
}

// ---------------------------------------------------------------------------
// Flash attention, causal, GQA (4 Q heads per KV head).
// Block = 4 waves, 64 q-rows (16 per wave), 64-key tiles staged in LDS.
// ---------------------------------------------------------------------------
#define ATT_BM 64
#define KV_BN  64

__global__ void __launch_bounds__(128)
attn_kernel(const bf16_t* __restrict__ Qb, const bf16_t* __restrict__ Kb,
            const bf16_t* __restrict__ Vb, bf16_t* __restrict__ Ob) {
    __shared__ __align__(16) bf16_t sK[KV_BN * HD];     // [j][d] row-major, 16KB
    __shared__ __align__(16) bf16_t sVt[HD * KV_BN];    // [d][j] transposed, 16KB
    __shared__ __align__(16) bf16_t sP[4][16 * KV_BN];  // per-wave P tile, 8KB

    const int h    = blockIdx.y;
    const int kvh  = h >> 2;                 // NH/NKV = 4
    const int q0   = blockIdx.x * ATT_BM;
    const int wave = threadIdx.x >> 5;
    const int lane = threadIdx.x & 31;
    const int qrow = q0 + wave * 16;
    const int n    = lane & 15;
    const int hm   = (lane >> 4) * 8;

    // Q fragments for this wave's 16 rows (d split into 4 K32 chunks)
    v16bf qfrag[4];
#pragma unroll
    for (int kc = 0; kc < 4; ++kc)
        qfrag[kc] = load_frag_a(Qb + h * HD, NH * HD, qrow, kc * 32);

    v8f o[8];
    float mrow[8], lsum[8];
#pragma unroll
    for (int dt = 0; dt < 8; ++dt) o[dt] = vzero8();
#pragma unroll
    for (int r = 0; r < 8; ++r) { mrow[r] = -3.0e38f; lsum[r] = 0.0f; }

    const int t  = threadIdx.x;
    const int lj = t >> 1;            // 0..63 : key row this thread stages
    const int d0 = (t & 1) * 64;      // half of head_dim

    for (int kb = 0; kb < q0 + ATT_BM; kb += KV_BN) {
        // ---- stage K tile via async global->LDS, V tile transposed ----
        const bf16_t* kg = Kb + (size_t)(kb + lj) * (NKV * HD) + kvh * HD + d0;
#pragma unroll
        for (int c = 0; c < 8; ++c)
            async_load_b128(lds_off(&sK[lj * HD + d0 + c * 8]), kg + c * 8);

        const bf16_t* vg = Vb + (size_t)(kb + lj) * (NKV * HD) + kvh * HD + d0;
#pragma unroll
        for (int c = 0; c < 64; ++c)
            sVt[(d0 + c) * KV_BN + lj] = vg[c];

        // warm caches for the next key block while we compute this one
        if (kb + KV_BN < q0 + ATT_BM) {
            __builtin_prefetch(kg + (size_t)KV_BN * (NKV * HD), 0, 1);
            __builtin_prefetch(vg + (size_t)KV_BN * (NKV * HD), 0, 1);
        }

        wait_async0();
        __syncthreads();

        // ---- S = Q K^T (4 jt tiles of 16 keys) ----
        v8f s[4];
#pragma unroll
        for (int jt = 0; jt < 4; ++jt) {
            s[jt] = vzero8();
#pragma unroll
            for (int kc = 0; kc < 4; ++kc) {
                v16bf kf = load_frag_b(sK, HD, jt * 16, kc * 32);
                s[jt] = WMMA_BF16(qfrag[kc], kf, s[jt]);
            }
        }

        // ---- scale + causal mask ----
        const float scale = 0.08838834764831845f;  // 1/sqrt(128)
#pragma unroll
        for (int jt = 0; jt < 4; ++jt)
#pragma unroll
            for (int r = 0; r < 8; ++r) {
                float val = s[jt][r] * scale;
                int kg_i = kb + jt * 16 + n;
                int qg_i = qrow + hm + r;
                s[jt][r] = (kg_i > qg_i) ? -1.0e30f : val;
            }

        // ---- online softmax stats (reduce over 16 lanes within a half) ----
        float alpha[8];
#pragma unroll
        for (int r = 0; r < 8; ++r) {
            float rowmax = -3.0e38f;
#pragma unroll
            for (int jt = 0; jt < 4; ++jt) rowmax = fmaxf(rowmax, s[jt][r]);
#pragma unroll
            for (int m = 1; m < 16; m <<= 1)
                rowmax = fmaxf(rowmax, __shfl_xor(rowmax, m, 32));
            float mn = fmaxf(mrow[r], rowmax);
            alpha[r] = __expf(mrow[r] - mn);
            mrow[r] = mn;
        }
#pragma unroll
        for (int r = 0; r < 8; ++r) {
            float rs = 0.0f;
#pragma unroll
            for (int jt = 0; jt < 4; ++jt) {
                float p = __expf(s[jt][r] - mrow[r]);
                s[jt][r] = p;
                rs += p;
            }
#pragma unroll
            for (int m = 1; m < 16; m <<= 1) rs += __shfl_xor(rs, m, 32);
            lsum[r] = lsum[r] * alpha[r] + rs;
        }
#pragma unroll
        for (int dt = 0; dt < 8; ++dt)
#pragma unroll
            for (int r = 0; r < 8; ++r) o[dt][r] *= alpha[r];

        // ---- P (C-layout) -> per-wave LDS tile (A-layout source) ----
#pragma unroll
        for (int jt = 0; jt < 4; ++jt)
#pragma unroll
            for (int r = 0; r < 8; ++r)
                sP[wave][(hm + r) * KV_BN + jt * 16 + n] = (bf16_t)s[jt][r];

        // ---- O += P V  (8 output d-tiles, 2 K32 chunks over the 64 keys) ----
#pragma unroll
        for (int kc2 = 0; kc2 < 2; ++kc2) {
            v16bf pf = load_frag_a(sP[wave], KV_BN, 0, kc2 * 32);
#pragma unroll
            for (int dt = 0; dt < 8; ++dt) {
                v16bf vf = load_frag_b(sVt, KV_BN, dt * 16, kc2 * 32);
                o[dt] = WMMA_BF16(pf, vf, o[dt]);
            }
        }
        __syncthreads();
    }

    // ---- epilogue: normalize, store bf16 ----
#pragma unroll
    for (int dt = 0; dt < 8; ++dt)
#pragma unroll
        for (int r = 0; r < 8; ++r)
            Ob[(size_t)(qrow + hm + r) * (NH * HD) + h * HD + dt * 16 + n] =
                (bf16_t)(o[dt][r] / lsum[r]);
}

// ---------------------------------------------------------------------------
// launch
// ---------------------------------------------------------------------------
extern "C" void kernel_launch(void* const* d_in, const int* in_sizes, int n_in,
                              void* d_out, int out_size, void* d_ws, size_t ws_size,
                              hipStream_t stream) {
    const float* x   = (const float*)d_in[0];
    const int* pos   = (const int*)d_in[1];
    const float* Wq  = (const float*)d_in[2];
    const float* Wk  = (const float*)d_in[3];
    const float* Wv  = (const float*)d_in[4];
    const float* Wo  = (const float*)d_in[5];
    const float* qnw = (const float*)d_in[6];
    const float* knw = (const float*)d_in[7];
    float* out = (float*)d_out;

    // workspace layout
    char* ws = (char*)d_ws;
    size_t off = 0;
    auto carve = [&](size_t bytes) { char* p = ws + off; off += (bytes + 255) & ~(size_t)255; return p; };
    bf16_t* xb     = (bf16_t*)carve((size_t)T_SEQ * HIDDEN * 2);
    bf16_t* wq_b   = (bf16_t*)carve((size_t)NH * HD * HIDDEN * 2);
    bf16_t* wk_b   = (bf16_t*)carve((size_t)NKV * HD * HIDDEN * 2);
    bf16_t* wv_b   = (bf16_t*)carve((size_t)NKV * HD * HIDDEN * 2);
    bf16_t* wo_b   = (bf16_t*)carve((size_t)HIDDEN * NH * HD * 2);
    float*  q_f    = (float*)carve((size_t)T_SEQ * NH * HD * 4);
    float*  k_f    = (float*)carve((size_t)T_SEQ * NKV * HD * 4);
    float*  v_f    = (float*)carve((size_t)T_SEQ * NKV * HD * 4);
    bf16_t* q_bf   = (bf16_t*)carve((size_t)T_SEQ * NH * HD * 2);
    bf16_t* k_bf   = (bf16_t*)carve((size_t)T_SEQ * NKV * HD * 2);
    bf16_t* v_bf   = (bf16_t*)carve((size_t)T_SEQ * NKV * HD * 2);
    bf16_t* att_bf = (bf16_t*)carve((size_t)T_SEQ * NH * HD * 2);

    auto cvt = [&](const float* src, bf16_t* dst, size_t n) {
        int n4 = (int)(n / 4);
        cvt_f32_to_bf16<<<(n4 + 255) / 256, 256, 0, stream>>>(src, dst, n4);
    };

    // 1) casts
    cvt(x,  xb,   (size_t)T_SEQ * HIDDEN);
    cvt(Wq, wq_b, (size_t)NH * HD * HIDDEN);
    cvt(Wk, wk_b, (size_t)NKV * HD * HIDDEN);
    cvt(Wv, wv_b, (size_t)NKV * HD * HIDDEN);
    cvt(Wo, wo_b, (size_t)HIDDEN * NH * HD);

    // 2) QKV projections (y = x W^T)
    gemm_bf16_nt<<<dim3((NH * HD) / BN, T_SEQ / BM), 256, 0, stream>>>(
        xb, wq_b, q_f, T_SEQ, NH * HD, HIDDEN);
    gemm_bf16_nt<<<dim3((NKV * HD) / BN, T_SEQ / BM), 256, 0, stream>>>(
        xb, wk_b, k_f, T_SEQ, NKV * HD, HIDDEN);
    gemm_bf16_nt<<<dim3((NKV * HD) / BN, T_SEQ / BM), 256, 0, stream>>>(
        xb, wv_b, v_f, T_SEQ, NKV * HD, HIDDEN);

    // 3) QK-norm + RoPE -> bf16 ; V -> bf16
    rmsnorm_rope_kernel<<<(T_SEQ * NH) / 8, 256, 0, stream>>>(
        q_f, qnw, pos, q_bf, NH, T_SEQ * NH);
    rmsnorm_rope_kernel<<<(T_SEQ * NKV) / 8, 256, 0, stream>>>(
        k_f, knw, pos, k_bf, NKV, T_SEQ * NKV);
    cvt(v_f, v_bf, (size_t)T_SEQ * NKV * HD);

    // 4) flash attention (causal, GQA)
    attn_kernel<<<dim3(T_SEQ / ATT_BM, NH), 128, 0, stream>>>(
        q_bf, k_bf, v_bf, att_bf);

    // 5) output projection -> fp32 d_out
    gemm_bf16_nt<<<dim3(HIDDEN / BN, T_SEQ / BM), 256, 0, stream>>>(
        att_bf, wo_b, out, T_SEQ, HIDDEN, NH * HD);
}
